// KNNImputationLayer_23742579212946
// MI455X (gfx1250) — matrix operand order repaired and checked by reading, not compiled, not run on hardware
//
#include <hip/hip_runtime.h>
#include <math.h>

#define N 4096
#define D 256
#define IWAVES 4

typedef __attribute__((ext_vector_type(16))) __bf16 v16bf;
typedef __attribute__((ext_vector_type(8)))  float  v8f;

typedef __attribute__((address_space(3))) void as3_void;

// ---------------- workspace layout (bytes) ----------------
// dist2 : N*N f32        64 MB   @ 0
// sqb   : N*D bf16        2 MB
// vb    : N*D bf16        2 MB
// xm2b  : N*D bf16        2 MB   (-2*xz, folds the -2 into the GEMM operand)
// xzb   : N*D bf16        2 MB
// vbits : D*(N/32) u32  128 KB   (column-major packed validity bits)
static const size_t OFF_DIST = 0;
static const size_t OFF_SQB  = (size_t)N * N * 4;
static const size_t OFF_VB   = OFF_SQB + (size_t)N * D * 2;
static const size_t OFF_XM2  = OFF_VB  + (size_t)N * D * 2;
static const size_t OFF_XZB  = OFF_XM2 + (size_t)N * D * 2;
static const size_t OFF_BITS = OFF_XZB + (size_t)N * D * 2;

// ---------------- phase 0: build bf16 operand matrices ----------------
__global__ void prep_kernel(const float* __restrict__ x, const int* __restrict__ mask,
                            __bf16* __restrict__ sqb, __bf16* __restrict__ vb,
                            __bf16* __restrict__ xm2b, __bf16* __restrict__ xzb) {
  int idx = blockIdx.x * blockDim.x + threadIdx.x;
  if (idx >= N * D) return;
  float xv = x[idx];
  bool  m  = mask[idx] != 0;
  float xz = m ? 0.0f : xv;
  float v  = m ? 0.0f : 1.0f;
  xzb[idx]  = (__bf16)xz;
  sqb[idx]  = (__bf16)(xz * xz);
  vb[idx]   = (__bf16)v;
  xm2b[idx] = (__bf16)(-2.0f * xz);
}

// ---------------- phase 0b: column-major packed validity bits ----------------
__global__ void bits_kernel(const int* __restrict__ mask, unsigned int* __restrict__ vbits) {
  int t = blockIdx.x * blockDim.x + threadIdx.x;     // t < D * (N/32)
  if (t >= D * (N / 32)) return;
  int j = t / (N / 32);
  int c = t % (N / 32);
  unsigned int bits = 0u;
  #pragma unroll 8
  for (int b = 0; b < 32; ++b) {
    int d = c * 32 + b;
    if (mask[(size_t)d * D + j] == 0) bits |= (1u << b);   // 1 = donor observes column j
  }
  vbits[t] = bits;
}

// A-matrix 16x32 bf16 fragment (ISA 7.12.2): lane&15 = row M; lanes 0-15 hold
// K = {0..7, 16..23}, lanes 16-31 hold K = {8..15, 24..31}  -> two 16B loads.
__device__ __forceinline__ v16bf ldfragA(const __bf16* __restrict__ base, int row0, int kt,
                                         int r, int hi) {
  const __bf16* p = base + (size_t)(row0 + r) * D + kt * 32 + hi * 8;
  union { v16bf f; uint4 q[2]; } u;
  u.q[0] = *(const uint4*)(p);
  u.q[1] = *(const uint4*)(p + 16);
  return u.f;
}

// B-matrix 32x16 bf16 fragment: lane&15 = column N; lanes 0-15 hold
// K = {0..7, 16..23}, lanes 16-31 hold K = {8..15, 24..31} -> two 16B loads.
// (B = xz^T, so lane n reads row n of the source matrix.)
__device__ __forceinline__ v16bf ldfragB(const __bf16* __restrict__ base, int col0, int kt,
                                         int r, int hi) {
  const __bf16* p = base + (size_t)(col0 + r) * D + kt * 32 + hi * 8;
  union { v16bf f; uint4 q[2]; } u;
  u.q[0] = *(const uint4*)(p);
  u.q[1] = *(const uint4*)(p + 16);
  return u.f;
}

// ---------------- phase 1: fused nan-euclidean distance GEMM ----------------
// Block = 256 threads (8 waves). Block tile = 128 rows x 64 cols.
// Per K-step, per sub-tile: d2 += sq.vT + v.sqT + (-2xz).xzT ; ns += v.vT.
__global__ __launch_bounds__(256) void gemm_kernel(const __bf16* __restrict__ sqb,
                                                   const __bf16* __restrict__ vb,
                                                   const __bf16* __restrict__ xm2b,
                                                   const __bf16* __restrict__ xzb,
                                                   float* __restrict__ dist2) {
  const int lane = (int)(threadIdx.x & 31u);
  const int wave = (int)(threadIdx.x >> 5);
  const int row0 = blockIdx.y * 128 + wave * 16;
  const int col0 = blockIdx.x * 64;
  const int r  = lane & 15;
  const int hi = lane >> 4;

  v8f accD0 = {}, accD1 = {}, accD2 = {}, accD3 = {};
  v8f accS0 = {}, accS1 = {}, accS2 = {}, accS3 = {};

  for (int kt = 0; kt < D / 32; ++kt) {
    if (kt + 1 < D / 32) {   // WGP-scope prefetch of next K-slab of the A rows
      __builtin_prefetch(sqb  + (size_t)(row0 + r) * D + (kt + 1) * 32 + hi * 8, 0, 3);
      __builtin_prefetch(xm2b + (size_t)(row0 + r) * D + (kt + 1) * 32 + hi * 8, 0, 3);
    }
    v16bf aSq = ldfragA(sqb , row0, kt, r, hi);
    v16bf aV  = ldfragA(vb  , row0, kt, r, hi);
    v16bf aX  = ldfragA(xm2b, row0, kt, r, hi);

    #pragma unroll
    for (int t = 0; t < 4; ++t) {
      const int c0 = col0 + t * 16;
      v16bf bV  = ldfragB(vb , c0, kt, r, hi);
      v16bf bSq = ldfragB(sqb, c0, kt, r, hi);
      v16bf bX  = ldfragB(xzb, c0, kt, r, hi);
      v8f* aD = (t == 0) ? &accD0 : (t == 1) ? &accD1 : (t == 2) ? &accD2 : &accD3;
      v8f* aS = (t == 0) ? &accS0 : (t == 1) ? &accS1 : (t == 2) ? &accS2 : &accS3;
      *aD = __builtin_amdgcn_wmma_f32_16x16x32_bf16(false, aSq, false, bV , (short)0, *aD, false, false);
      *aD = __builtin_amdgcn_wmma_f32_16x16x32_bf16(false, aV , false, bSq, (short)0, *aD, false, false);
      *aD = __builtin_amdgcn_wmma_f32_16x16x32_bf16(false, aX , false, bX , (short)0, *aD, false, false);
      *aS = __builtin_amdgcn_wmma_f32_16x16x32_bf16(false, aV , false, bV , (short)0, *aS, false, false);
    }
  }

  // epilogue: dist2 = n_shared>0 ? max(d2,0) * D/n_shared : +inf
  const float INF = __builtin_inff();
  #pragma unroll
  for (int t = 0; t < 4; ++t) {
    v8f dA = (t == 0) ? accD0 : (t == 1) ? accD1 : (t == 2) ? accD2 : accD3;
    v8f sA = (t == 0) ? accS0 : (t == 1) ? accS1 : (t == 2) ? accS2 : accS3;
    #pragma unroll
    for (int e = 0; e < 8; ++e) {
      int mrow = row0 + e + 8 * hi;                 // C/D layout: lanes 16-31 hold M=8..15
      int ncol = col0 + t * 16 + r;
      float d2 = fmaxf(dA[e], 0.0f);
      float ns = sA[e];
      float dv = (ns > 0.5f) ? d2 * ((float)D / ns) : INF;
      dist2[(size_t)mrow * N + ncol] = dv;
    }
  }
}

// async 16B global -> LDS copy (ASYNCcnt-tracked); inline asm per ISA §15.18:
//   global_load_async_to_lds_b128 vdst(lds byte addr), vaddr(64b global), off
__device__ __forceinline__ void async_copy16(const float4* gsrc, float4* ldst) {
  unsigned lds_off = (unsigned)(unsigned long long)(as3_void*)ldst;  // 32-bit LDS addr
  asm volatile("global_load_async_to_lds_b128 %0, %1, off"
               :: "v"(lds_off), "v"(gsrc)
               : "memory");
}

// per-lane sorted top-5 insertion (ascending); strict < keeps earliest index on ties
__device__ __forceinline__ void insert5(float d, int c,
    float& t0, float& t1, float& t2, float& t3, float& t4,
    int& i0, int& i1, int& i2, int& i3, int& i4) {
  if (d < t4) {
    if (d < t0)      { t4=t3;i4=i3; t3=t2;i3=i2; t2=t1;i2=i1; t1=t0;i1=i0; t0=d;i0=c; }
    else if (d < t1) { t4=t3;i4=i3; t3=t2;i3=i2; t2=t1;i2=i1; t1=d;i1=c; }
    else if (d < t2) { t4=t3;i4=i3; t3=t2;i3=i2; t2=d;i2=c; }
    else if (d < t3) { t4=t3;i4=i3; t3=d;i3=c; }
    else             { t4=d;i4=c; }
  }
}

// ---------------- phase 2: top-5 donor selection + impute ----------------
// One wave per row; dist2 row staged into LDS via async global->LDS copies.
__global__ __launch_bounds__(IWAVES * 32) void impute_kernel(
    const float* __restrict__ x, const int* __restrict__ mask,
    const float* __restrict__ dist2, const unsigned int* __restrict__ vbits,
    float* __restrict__ out) {
  __shared__ float lds[IWAVES][N];            // 4 * 16 KB = 64 KB
  const int lane = (int)(threadIdx.x & 31u);
  const int wave = (int)(threadIdx.x >> 5);
  const int i = blockIdx.x * IWAVES + wave;
  float* srow = lds[wave];
  const float INF = __builtin_inff();

  // stage this row's distances into LDS (async DMA path, no VGPR round-trip)
  const float4* drow = (const float4*)(dist2 + (size_t)i * N);
  float4* s4 = (float4*)srow;
  for (int c = lane; c < N / 4; c += 32)
    async_copy16(drow + c, s4 + c);

  // pass 1: copy observed entries straight through (overlaps the async copy)
  const float* xrow = x + (size_t)i * D;
  const int*   mrow = mask + (size_t)i * D;
  float*       orow = out + (size_t)i * D;
  for (int j = lane; j < D; j += 32)
    if (!mrow[j]) orow[j] = xrow[j];

  asm volatile("s_wait_asynccnt 0" ::: "memory");
  __syncthreads();

  // pass 2: impute each missing column
  for (int j = 0; j < D; ++j) {
    if (!mrow[j]) continue;                   // wave-uniform branch

    float t0 = INF, t1 = INF, t2 = INF, t3 = INF, t4 = INF;
    int   i0 = 0x7fffffff, i1 = 0x7fffffff, i2 = 0x7fffffff, i3 = 0x7fffffff, i4 = 0x7fffffff;
    const unsigned int* vb = vbits + (size_t)j * (N / 32);

    for (int it = 0; it < N / 32; ++it) {
      unsigned int bits = vb[it];             // wave-uniform dword, bit = lane's candidate
      int c = it * 32 + lane;
      float dcand = ((bits >> lane) & 1u) ? srow[c] : INF;
      insert5(dcand, c, t0, t1, t2, t3, t4, i0, i1, i2, i3, i4);
    }

    // 5 rounds of wave-wide argmin; pop winner from owner lane's sorted list
    float sumw = 0.0f, sumv = 0.0f;
    #pragma unroll
    for (int s = 0; s < 5; ++s) {
      float d = t0; int id = i0;
      #pragma unroll
      for (int off = 16; off; off >>= 1) {
        float od = __shfl_xor(d, off, 32);
        int   oi = __shfl_xor(id, off, 32);
        if (od < d || (od == d && oi < id)) { d = od; id = oi; }
      }
      if (t0 == d && i0 == id) {              // pop (idx is unique among finite entries)
        t0 = t1; i0 = i1; t1 = t2; i1 = i2; t2 = t3; i2 = i3; t3 = t4; i3 = i4;
        t4 = INF; i4 = 0x7fffffff;
      }
      if (d < INF) {                          // w = isfinite(dist)
        sumw += 1.0f;
        sumv += x[(size_t)id * D + j];        // broadcast gather of donor value
      }
    }
    float imp = sumv / fmaxf(sumw, 1.0f);
    if (lane == 0) orow[j] = imp;
  }
}

extern "C" void kernel_launch(void* const* d_in, const int* in_sizes, int n_in,
                              void* d_out, int out_size, void* d_ws, size_t ws_size,
                              hipStream_t stream) {
  const float* x    = (const float*)d_in[0];
  const int*   mask = (const int*)d_in[1];   // jax bool -> int per harness contract
  (void)in_sizes; (void)n_in; (void)out_size; (void)ws_size; // k fixed at 5
  float* out = (float*)d_out;

  char* ws = (char*)d_ws;
  float*        dist2 = (float*)(ws + OFF_DIST);
  __bf16*       sqb   = (__bf16*)(ws + OFF_SQB);
  __bf16*       vb    = (__bf16*)(ws + OFF_VB);
  __bf16*       xm2b  = (__bf16*)(ws + OFF_XM2);
  __bf16*       xzb   = (__bf16*)(ws + OFF_XZB);
  unsigned int* vbits = (unsigned int*)(ws + OFF_BITS);

  prep_kernel<<<(N * D + 255) / 256, 256, 0, stream>>>(x, mask, sqb, vb, xm2b, xzb);
  bits_kernel<<<(D * (N / 32) + 255) / 256, 256, 0, stream>>>(mask, vbits);
  gemm_kernel<<<dim3(N / 64, N / 128), 256, 0, stream>>>(sqb, vb, xm2b, xzb, dist2);
  impute_kernel<<<N / IWAVES, IWAVES * 32, 0, stream>>>(x, mask, dist2, vbits, out);
}
